// FastGraphGPS_53463752900683
// MI455X (gfx1250) — matrix operand order repaired and checked by reading, not compiled, not run on hardware
//
#include <hip/hip_runtime.h>
#include <hip/hip_bf16.h>
#include <math.h>
#include <limits.h>

typedef __attribute__((ext_vector_type(16))) __bf16 v16bf;
typedef __attribute__((ext_vector_type(8)))  float  v8f;

#define NNODE 4096
#define NEDGE 131072
#define INCH  512
#define CDIM  256
#define NHEAD 4
#define DHEAD 64
#define NLAYER 3
#define NCLSS 2
#define ETOT (NEDGE + NNODE)
#define HC   (NHEAD * CDIM) /* 1024 */

// ---------------------------------------------------------------- async-LDS path
#if defined(__AMDGCN__) && __has_builtin(__builtin_amdgcn_global_load_async_to_lds_b128) && \
    __has_builtin(__builtin_amdgcn_s_wait_asynccnt)
#define HAS_ASYNC_LDS 1
#else
#define HAS_ASYNC_LDS 0
#endif

#if HAS_ASYNC_LDS
typedef int v4i_ __attribute__((vector_size(4 * sizeof(int))));
typedef __attribute__((address_space(1))) v4i_* glob_v4i;
typedef __attribute__((address_space(3))) v4i_* lds_v4i;
// generic->LDS: low 32 bits of a generic LDS pointer are the LDS offset
__device__ __forceinline__ lds_v4i to_lds(const void* p) {
    return (lds_v4i)(unsigned int)(unsigned long long)p;
}
__device__ __forceinline__ glob_v4i to_glob(const void* p) {
    return (glob_v4i)(unsigned long long)p;
}
#endif

// ---------------------------------------------------------------- helpers
__device__ __forceinline__ float gelu_erf(float x) {
    return 0.5f * x * (1.0f + erff(x * 0.70710678118654752f));
}
// monotone float<->int encoding for atomicMax on floats
__device__ __forceinline__ int   encf(float f) { int b = __float_as_int(f); return b >= 0 ? b : (b ^ 0x7FFFFFFF); }
__device__ __forceinline__ float decf(int k)   { return __int_as_float(k >= 0 ? k : (k ^ 0x7FFFFFFF)); }

// ---------------------------------------------------------------- fp32 -> bf16
__global__ void f2bf_kernel(const float* __restrict__ in, __bf16* __restrict__ out, int n) {
    int i = blockIdx.x * blockDim.x + threadIdx.x;
    if (i < n) out[i] = (__bf16)in[i];
}

// ---------------------------------------------------------------- tiled WMMA GEMM
// out[M,Nd] = A[M,K] @ W[Nd,K]^T (+bias) (+gelu). M=4096; Nd%64==0, K%32==0.
// Block tile 128x64, 8 waves, each wave owns a 32x32 slab (4 accumulators).
__global__ __launch_bounds__(256) void gemm_bf16(
    const __bf16* __restrict__ A, const __bf16* __restrict__ W,
    const float* __restrict__ bias, float* __restrict__ outF,
    __bf16* __restrict__ outB, int Ndim, int K, int act)
{
    __shared__ __bf16 sA[128][32];
    __shared__ __bf16 sW[64][32];
    const int tid  = threadIdx.x;
    const int lane = tid & 31, wave = tid >> 5;
    const int grp  = lane >> 4, lr = lane & 15;
    const int mw   = wave & 3,  nw = wave >> 2;          // 4 (M) x 2 (N) waves
    const int m0   = blockIdx.x * 128, n0 = blockIdx.y * 64;
    const int ra   = tid >> 1, ca = (tid & 1) * 16;      // A loader: 128 rows x 2 halves
    const int rw   = tid >> 2, cw = (tid & 3) * 8;       // W loader: 64 rows x 4 octets

    v8f acc00 = {}, acc01 = {}, acc10 = {}, acc11 = {};
    for (int k0 = 0; k0 < K; k0 += 32) {
        const __bf16* gA = &A[(size_t)(m0 + ra) * K + k0 + ca];
        const __bf16* gW = &W[(size_t)(n0 + rw) * K + k0 + cw];
#if HAS_ASYNC_LDS
        __builtin_amdgcn_global_load_async_to_lds_b128(to_glob(gA),     to_lds(&sA[ra][ca]),     0, 0);
        __builtin_amdgcn_global_load_async_to_lds_b128(to_glob(gA + 8), to_lds(&sA[ra][ca + 8]), 0, 0);
        __builtin_amdgcn_global_load_async_to_lds_b128(to_glob(gW),     to_lds(&sW[rw][cw]),     0, 0);
#else
        *(uint4*)(&sA[ra][ca])     = *(const uint4*)(gA);
        *(uint4*)(&sA[ra][ca + 8]) = *(const uint4*)(gA + 8);
        *(uint4*)(&sW[rw][cw])     = *(const uint4*)(gW);
#endif
        if (k0 + 32 < K) {
            __builtin_prefetch(gA + 32, 0, 3);
            __builtin_prefetch(gW + 32, 0, 3);
        }
#if HAS_ASYNC_LDS
        __builtin_amdgcn_s_wait_asynccnt(0);
#endif
        __syncthreads();
        v16bf a0, a1, b0, b1;
#pragma unroll
        for (int e = 0; e < 16; ++e) {
            int ka = 16 * (e >> 3) + 8 * grp + (e & 7);
            a0[e] = sA[mw * 32 + lr][ka];
            a1[e] = sA[mw * 32 + 16 + lr][ka];
        }
#pragma unroll
        for (int e = 0; e < 16; ++e) {
            int kb = 16 * (e >> 3) + 8 * grp + (e & 7);
            b0[e] = sW[nw * 32 + lr][kb];
            b1[e] = sW[nw * 32 + 16 + lr][kb];
        }
        acc00 = __builtin_amdgcn_wmma_f32_16x16x32_bf16(false, a0, false, b0, (short)0, acc00, false, false);
        acc01 = __builtin_amdgcn_wmma_f32_16x16x32_bf16(false, a0, false, b1, (short)0, acc01, false, false);
        acc10 = __builtin_amdgcn_wmma_f32_16x16x32_bf16(false, a1, false, b0, (short)0, acc10, false, false);
        acc11 = __builtin_amdgcn_wmma_f32_16x16x32_bf16(false, a1, false, b1, (short)0, acc11, false, false);
        __syncthreads();
    }
#pragma unroll
    for (int v = 0; v < 8; ++v) {
        int row0 = m0 + mw * 32 + v + 8 * grp;
        int row1 = row0 + 16;
        int c0   = n0 + nw * 32 + lr;
        int c1   = c0 + 16;
        float x00 = acc00[v], x01 = acc01[v], x10 = acc10[v], x11 = acc11[v];
        if (bias) {
            float bc0 = bias[c0], bc1 = bias[c1];
            x00 += bc0; x01 += bc1; x10 += bc0; x11 += bc1;
        }
        if (act) { x00 = gelu_erf(x00); x01 = gelu_erf(x01); x10 = gelu_erf(x10); x11 = gelu_erf(x11); }
        if (outF) {
            outF[(size_t)row0 * Ndim + c0] = x00; outF[(size_t)row0 * Ndim + c1] = x01;
            outF[(size_t)row1 * Ndim + c0] = x10; outF[(size_t)row1 * Ndim + c1] = x11;
        }
        if (outB) {
            outB[(size_t)row0 * Ndim + c0] = (__bf16)x00; outB[(size_t)row0 * Ndim + c1] = (__bf16)x01;
            outB[(size_t)row1 * Ndim + c0] = (__bf16)x10; outB[(size_t)row1 * Ndim + c1] = (__bf16)x11;
        }
    }
}

// ---------------------------------------------------------------- flash attention
// qkv: [N, 3C] bf16 (q|k|v per head dh=64). One wave per (16 q-rows, head).
__global__ __launch_bounds__(32) void flash_attn(const __bf16* __restrict__ qkv,
                                                 __bf16* __restrict__ out)
{
    __shared__ __bf16 P[16 * 32];
    const int lane = threadIdx.x;
    const int grp  = lane >> 4, lr = lane & 15;
    const int hid  = blockIdx.y;
    const int q0   = blockIdx.x * 16;
    const int S3C  = 3 * CDIM;
    const int qoff = hid * DHEAD;
    const int koff = CDIM + hid * DHEAD;
    const int voff = 2 * CDIM + hid * DHEAD;
    const float qs = 0.125f; // 1/sqrt(64), folded into Q

    v16bf qa0, qa1;
#pragma unroll
    for (int e = 0; e < 16; ++e) {
        int d = 16 * grp + e;
        qa0[e] = (__bf16)(qs * (float)qkv[(size_t)(q0 + lr) * S3C + qoff + d]);
        qa1[e] = (__bf16)(qs * (float)qkv[(size_t)(q0 + lr) * S3C + qoff + 32 + d]);
    }

    float mrow[8], lrow[8], scv[8];
    v8f o0 = {}, o1 = {}, o2 = {}, o3 = {};
#pragma unroll
    for (int v = 0; v < 8; ++v) { mrow[v] = -3.0e38f; lrow[v] = 0.f; }

    for (int kt = 0; kt < NNODE; kt += 32) {
        v8f s0 = {}, s1 = {};
        {
            v16bf kb;
#pragma unroll
            for (int e = 0; e < 16; ++e) kb[e] = qkv[(size_t)(kt + lr) * S3C + koff + 16 * grp + e];
            s0 = __builtin_amdgcn_wmma_f32_16x16x32_bf16(false, qa0, false, kb, (short)0, s0, false, false);
#pragma unroll
            for (int e = 0; e < 16; ++e) kb[e] = qkv[(size_t)(kt + lr) * S3C + koff + 32 + 16 * grp + e];
            s0 = __builtin_amdgcn_wmma_f32_16x16x32_bf16(false, qa1, false, kb, (short)0, s0, false, false);
#pragma unroll
            for (int e = 0; e < 16; ++e) kb[e] = qkv[(size_t)(kt + 16 + lr) * S3C + koff + 16 * grp + e];
            s1 = __builtin_amdgcn_wmma_f32_16x16x32_bf16(false, qa0, false, kb, (short)0, s1, false, false);
#pragma unroll
            for (int e = 0; e < 16; ++e) kb[e] = qkv[(size_t)(kt + 16 + lr) * S3C + koff + 32 + 16 * grp + e];
            s1 = __builtin_amdgcn_wmma_f32_16x16x32_bf16(false, qa1, false, kb, (short)0, s1, false, false);
        }
#pragma unroll
        for (int v = 0; v < 8; ++v) {
            float t = fmaxf(s0[v], s1[v]);
#pragma unroll
            for (int off = 1; off < 16; off <<= 1) t = fmaxf(t, __shfl_xor(t, off, 16));
            float nm = fmaxf(mrow[v], t);
            float sc = __expf(mrow[v] - nm);
            mrow[v]  = nm;
            float p0 = __expf(s0[v] - nm);
            float p1 = __expf(s1[v] - nm);
            float rs = p0 + p1;
#pragma unroll
            for (int off = 1; off < 16; off <<= 1) rs += __shfl_xor(rs, off, 16);
            lrow[v] = lrow[v] * sc + rs;
            scv[v]  = sc;
            P[(v + 8 * grp) * 32 + lr]      = (__bf16)p0;
            P[(v + 8 * grp) * 32 + 16 + lr] = (__bf16)p1;
        }
        __syncthreads();
        v16bf pa;
#pragma unroll
        for (int e = 0; e < 16; ++e)
            pa[e] = P[lr * 32 + 16 * grp + e];
#pragma unroll
        for (int v = 0; v < 8; ++v) { o0[v] *= scv[v]; o1[v] *= scv[v]; o2[v] *= scv[v]; o3[v] *= scv[v]; }
        v16bf vb;
#pragma unroll
        for (int e = 0; e < 16; ++e) vb[e] = qkv[(size_t)(kt + 16 * grp + e) * S3C + voff + 0 + lr];
        o0 = __builtin_amdgcn_wmma_f32_16x16x32_bf16(false, pa, false, vb, (short)0, o0, false, false);
#pragma unroll
        for (int e = 0; e < 16; ++e) vb[e] = qkv[(size_t)(kt + 16 * grp + e) * S3C + voff + 16 + lr];
        o1 = __builtin_amdgcn_wmma_f32_16x16x32_bf16(false, pa, false, vb, (short)0, o1, false, false);
#pragma unroll
        for (int e = 0; e < 16; ++e) vb[e] = qkv[(size_t)(kt + 16 * grp + e) * S3C + voff + 32 + lr];
        o2 = __builtin_amdgcn_wmma_f32_16x16x32_bf16(false, pa, false, vb, (short)0, o2, false, false);
#pragma unroll
        for (int e = 0; e < 16; ++e) vb[e] = qkv[(size_t)(kt + 16 * grp + e) * S3C + voff + 48 + lr];
        o3 = __builtin_amdgcn_wmma_f32_16x16x32_bf16(false, pa, false, vb, (short)0, o3, false, false);
        __syncthreads();
    }
#pragma unroll
    for (int v = 0; v < 8; ++v) {
        int row = q0 + v + 8 * grp;
        float inv = 1.0f / lrow[v];
        out[(size_t)row * CDIM + hid * DHEAD + 0  + lr] = (__bf16)(o0[v] * inv);
        out[(size_t)row * CDIM + hid * DHEAD + 16 + lr] = (__bf16)(o1[v] * inv);
        out[(size_t)row * CDIM + hid * DHEAD + 32 + lr] = (__bf16)(o2[v] * inv);
        out[(size_t)row * CDIM + hid * DHEAD + 48 + lr] = (__bf16)(o3[v] * inv);
    }
}

// ---------------------------------------------------------------- GATv2 pieces
__global__ void gat_init(float* __restrict__ accum, float* __restrict__ denom, int* __restrict__ mint) {
    int idx = blockIdx.x * blockDim.x + threadIdx.x;
    if (idx < NNODE * HC) accum[idx] = 0.f;
    if (idx < NNODE * NHEAD) { denom[idx] = 0.f; mint[idx] = INT_MIN; }
}

__global__ __launch_bounds__(256) void gat_score(
    const float* __restrict__ xl, const float* __restrict__ xr,
    const int* __restrict__ ei, const float* __restrict__ att,
    float* __restrict__ score, int* __restrict__ mint)
{
    __shared__ float red[256];
    __shared__ int   sd[2];
    int e = blockIdx.x, t = threadIdx.x;
    if (t == 0) {
        if (e < NEDGE) { sd[0] = ei[e]; sd[1] = ei[NEDGE + e]; }
        else           { sd[0] = sd[1] = e - NEDGE; }
    }
    __syncthreads();
    int src = sd[0], dst = sd[1];
    float p[NHEAD];
#pragma unroll
    for (int h = 0; h < NHEAD; ++h) {
        int i = h * CDIM + t;
        float v = xl[(size_t)src * HC + i] + xr[(size_t)dst * HC + i];
        v = v > 0.f ? v : 0.2f * v;           // leaky_relu 0.2
        p[h] = v * att[i];
    }
    for (int h = 0; h < NHEAD; ++h) {
        red[t] = p[h]; __syncthreads();
        for (int s = 128; s > 0; s >>= 1) { if (t < s) red[t] += red[t + s]; __syncthreads(); }
        if (t == 0) {
            float sc = red[0];
            score[(size_t)e * NHEAD + h] = sc;
            atomicMax(&mint[dst * NHEAD + h], encf(sc));
        }
        __syncthreads();
    }
}

__global__ void gat_exp(float* __restrict__ score, const int* __restrict__ ei,
                        const int* __restrict__ mint, float* __restrict__ denom)
{
    int idx = blockIdx.x * blockDim.x + threadIdx.x;
    if (idx >= ETOT * NHEAD) return;
    int e = idx >> 2, h = idx & 3;
    int dst = (e < NEDGE) ? ei[NEDGE + e] : e - NEDGE;
    float m  = decf(mint[dst * NHEAD + h]);
    float ex = __expf(score[idx] - m);
    score[idx] = ex;
    atomicAdd(&denom[dst * NHEAD + h], ex);
}

__global__ __launch_bounds__(256) void gat_agg(
    const float* __restrict__ exs, const float* __restrict__ denom,
    const float* __restrict__ xl, const int* __restrict__ ei,
    float* __restrict__ accum)
{
    int e = blockIdx.x, t = threadIdx.x;
    int src, dst;
    if (e < NEDGE) { src = ei[e]; dst = ei[NEDGE + e]; } else { src = dst = e - NEDGE; }
#pragma unroll
    for (int h = 0; h < NHEAD; ++h) {
        float alpha = exs[(size_t)e * NHEAD + h] / denom[dst * NHEAD + h];
        int i = h * CDIM + t;
        atomicAdd(&accum[(size_t)dst * HC + i], alpha * xl[(size_t)src * HC + i]);
    }
}

// ---------------------------------------------------------------- residual + LN
__global__ __launch_bounds__(CDIM) void add_ln(
    const float* __restrict__ base, const float* __restrict__ delta,
    const float* __restrict__ accumH, const float* __restrict__ gatb,
    const float* __restrict__ g, const float* __restrict__ b,
    float* __restrict__ outF, __bf16* __restrict__ outB)
{
    __shared__ float red[CDIM];
    int n = blockIdx.x, c = threadIdx.x;
    float x = base[(size_t)n * CDIM + c];
    if (delta) x += delta[(size_t)n * CDIM + c];
    if (accumH) {
        float a = 0.f;
#pragma unroll
        for (int h = 0; h < NHEAD; ++h) a += accumH[(size_t)n * HC + h * CDIM + c];
        x += a * (1.0f / NHEAD) + gatb[c];
    }
    red[c] = x; __syncthreads();
    for (int s = CDIM / 2; s > 0; s >>= 1) { if (c < s) red[c] += red[c + s]; __syncthreads(); }
    float mu = red[0] * (1.0f / CDIM);
    __syncthreads();
    float dx = x - mu;
    red[c] = dx * dx; __syncthreads();
    for (int s = CDIM / 2; s > 0; s >>= 1) { if (c < s) red[c] += red[c + s]; __syncthreads(); }
    float var = red[0] * (1.0f / CDIM);
    float y = dx * rsqrtf(var + 1e-5f) * g[c] + b[c];
    outF[(size_t)n * CDIM + c] = y;
    outB[(size_t)n * CDIM + c] = (__bf16)y;
}

// ---------------------------------------------------------------- final 2-class head
__global__ void out_head(const __bf16* __restrict__ z, const float* __restrict__ w,
                         const float* __restrict__ b, float* __restrict__ out)
{
    int idx = blockIdx.x * blockDim.x + threadIdx.x;
    if (idx >= NNODE * NCLSS) return;
    int n = idx / NCLSS, cls = idx % NCLSS;
    float s = b[cls];
    for (int k = 0; k < CDIM; ++k) s += (float)z[(size_t)n * CDIM + k] * w[cls * CDIM + k];
    out[idx] = s;
}

// ================================================================ host
extern "C" void kernel_launch(void* const* d_in, const int* in_sizes, int n_in,
                              void* d_out, int out_size, void* d_ws, size_t ws_size,
                              hipStream_t stream)
{
    (void)in_sizes; (void)n_in; (void)out_size; (void)ws_size;
    const float* x        = (const float*)d_in[0];
    const int*   ei       = (const int*)  d_in[1];
    const float* in_w     = (const float*)d_in[2];
    const float* in_b     = (const float*)d_in[3];
    const float* gat_wl   = (const float*)d_in[4];
    const float* gat_wr   = (const float*)d_in[5];
    const float* gat_att  = (const float*)d_in[6];
    const float* gat_b    = (const float*)d_in[7];
    const float* mha_in_w = (const float*)d_in[8];
    const float* mha_in_b = (const float*)d_in[9];
    const float* mha_out_w= (const float*)d_in[10];
    const float* mha_out_b= (const float*)d_in[11];
    const float* ln1_g    = (const float*)d_in[12];
    const float* ln1_b    = (const float*)d_in[13];
    const float* ln2_g    = (const float*)d_in[14];
    const float* ln2_b    = (const float*)d_in[15];
    const float* ln3_g    = (const float*)d_in[16];
    const float* ln3_b    = (const float*)d_in[17];
    const float* ffn_w1   = (const float*)d_in[18];
    const float* ffn_b1   = (const float*)d_in[19];
    const float* ffn_w2   = (const float*)d_in[20];
    const float* ffn_b2   = (const float*)d_in[21];
    const float* out_w1   = (const float*)d_in[22];
    const float* out_b1   = (const float*)d_in[23];
    const float* out_w2   = (const float*)d_in[24];
    const float* out_b2   = (const float*)d_in[25];

    char* ws = (char*)d_ws;
    size_t off = 0;
    auto carve = [&](size_t bytes) -> void* {
        void* p = ws + off; off += (bytes + 255) & ~(size_t)255; return p;
    };
    float*  h_f    = (float*) carve((size_t)NNODE * CDIM * 4);
    __bf16* h_b    = (__bf16*)carve((size_t)NNODE * CDIM * 2);
    __bf16* x_b    = (__bf16*)carve((size_t)NNODE * INCH * 2);
    float*  xl     = (float*) carve((size_t)NNODE * HC * 4);
    float*  xr     = (float*) carve((size_t)NNODE * HC * 4);
    float*  accum  = (float*) carve((size_t)NNODE * HC * 4);
    float*  score  = (float*) carve((size_t)ETOT * NHEAD * 4);
    int*    mint   = (int*)   carve((size_t)NNODE * NHEAD * 4);
    float*  denom  = (float*) carve((size_t)NNODE * NHEAD * 4);
    __bf16* qkv_b  = (__bf16*)carve((size_t)NNODE * 3 * CDIM * 2);
    __bf16* attn_b = (__bf16*)carve((size_t)NNODE * CDIM * 2);
    float*  delta  = (float*) carve((size_t)NNODE * CDIM * 4);
    __bf16* ffn_hb = (__bf16*)carve((size_t)NNODE * 2 * CDIM * 2);
    __bf16* z_b    = (__bf16*)carve((size_t)NNODE * CDIM * 2);
    __bf16* inw_b  = (__bf16*)carve((size_t)CDIM * INCH * 2);
    __bf16* wl_b   = (__bf16*)carve((size_t)NLAYER * HC * CDIM * 2);
    __bf16* wr_b   = (__bf16*)carve((size_t)NLAYER * HC * CDIM * 2);
    __bf16* qw_b   = (__bf16*)carve((size_t)NLAYER * 3 * CDIM * CDIM * 2);
    __bf16* ow_b   = (__bf16*)carve((size_t)NLAYER * CDIM * CDIM * 2);
    __bf16* f1_b   = (__bf16*)carve((size_t)NLAYER * 2 * CDIM * CDIM * 2);
    __bf16* f2_b   = (__bf16*)carve((size_t)NLAYER * CDIM * 2 * CDIM * 2);
    __bf16* o1_b   = (__bf16*)carve((size_t)CDIM * CDIM * 2);

    auto conv = [&](const float* s, __bf16* d, int n) {
        f2bf_kernel<<<(n + 255) / 256, 256, 0, stream>>>(s, d, n);
    };
    conv(x,         x_b,  NNODE * INCH);
    conv(in_w,      inw_b, CDIM * INCH);
    conv(gat_wl,    wl_b, NLAYER * HC * CDIM);
    conv(gat_wr,    wr_b, NLAYER * HC * CDIM);
    conv(mha_in_w,  qw_b, NLAYER * 3 * CDIM * CDIM);
    conv(mha_out_w, ow_b, NLAYER * CDIM * CDIM);
    conv(ffn_w1,    f1_b, NLAYER * 2 * CDIM * CDIM);
    conv(ffn_w2,    f2_b, NLAYER * CDIM * 2 * CDIM);
    conv(out_w1,    o1_b, CDIM * CDIM);

    auto gemm = [&](const __bf16* Am, const __bf16* Wm, const float* bi,
                    float* oF, __bf16* oB, int Nd, int K, int act) {
        dim3 grid(NNODE / 128, Nd / 64);
        gemm_bf16<<<grid, 256, 0, stream>>>(Am, Wm, bi, oF, oB, Nd, K, act);
    };

    // input projection
    gemm(x_b, inw_b, in_b, h_f, h_b, CDIM, INCH, 0);

    for (int l = 0; l < NLAYER; ++l) {
        // --- GATv2 local branch ---
        gemm(h_b, wl_b + (size_t)l * HC * CDIM, nullptr, xl, nullptr, HC, CDIM, 0);
        gemm(h_b, wr_b + (size_t)l * HC * CDIM, nullptr, xr, nullptr, HC, CDIM, 0);
        gat_init<<<(NNODE * HC) / 256, 256, 0, stream>>>(accum, denom, mint);
        gat_score<<<ETOT, 256, 0, stream>>>(xl, xr, ei, gat_att + l * NHEAD * CDIM, score, mint);
        gat_exp<<<(ETOT * NHEAD + 255) / 256, 256, 0, stream>>>(score, ei, mint, denom);
        gat_agg<<<ETOT, 256, 0, stream>>>(score, denom, xl, ei, accum);
        add_ln<<<NNODE, CDIM, 0, stream>>>(h_f, nullptr, accum, gat_b + l * CDIM,
                                           ln1_g + l * CDIM, ln1_b + l * CDIM, h_f, h_b);
        // --- MHA global branch ---
        gemm(h_b, qw_b + (size_t)l * 3 * CDIM * CDIM, mha_in_b + l * 3 * CDIM,
             nullptr, qkv_b, 3 * CDIM, CDIM, 0);
        flash_attn<<<dim3(NNODE / 16, NHEAD), 32, 0, stream>>>(qkv_b, attn_b);
        gemm(attn_b, ow_b + (size_t)l * CDIM * CDIM, mha_out_b + l * CDIM,
             delta, nullptr, CDIM, CDIM, 0);
        add_ln<<<NNODE, CDIM, 0, stream>>>(h_f, delta, nullptr, nullptr,
                                           ln2_g + l * CDIM, ln2_b + l * CDIM, h_f, h_b);
        // --- FFN ---
        gemm(h_b, f1_b + (size_t)l * 2 * CDIM * CDIM, ffn_b1 + l * 2 * CDIM,
             nullptr, ffn_hb, 2 * CDIM, CDIM, 1);
        gemm(ffn_hb, f2_b + (size_t)l * CDIM * 2 * CDIM, ffn_b2 + l * CDIM,
             delta, nullptr, CDIM, 2 * CDIM, 0);
        add_ln<<<NNODE, CDIM, 0, stream>>>(h_f, delta, nullptr, nullptr,
                                           ln3_g + l * CDIM, ln3_b + l * CDIM, h_f, h_b);
    }

    // output head
    gemm(h_b, o1_b, out_b1, nullptr, z_b, CDIM, CDIM, 1);
    out_head<<<(NNODE * NCLSS + 255) / 256, 256, 0, stream>>>(z_b, out_w2, out_b2, (float*)d_out);
}
